// AttentionBlock_231928234030
// MI455X (gfx1250) — matrix operand order repaired and checked by reading, not compiled, not run on hardware
//
#include <hip/hip_runtime.h>
#include <hip/hip_bf16.h>
#include <cstdint>

typedef __bf16 bf16_t;
typedef __attribute__((ext_vector_type(16))) __bf16 v16bf;
typedef __attribute__((ext_vector_type(8)))  __bf16 v8bf;
typedef __attribute__((ext_vector_type(8)))  float  v8f;
typedef __attribute__((ext_vector_type(4)))  unsigned int v4u;
typedef __attribute__((ext_vector_type(8)))  int v8i;
typedef __attribute__((ext_vector_type(4)))  int v4i;

#define NB    16
#define CC    512
#define TT    1024
#define NHD   8
#define CH    64
#define QKV_C 1536

#if defined(__has_builtin)
#if __has_builtin(__builtin_amdgcn_tensor_load_to_lds)
#define HAVE_TDM 1
#endif
#endif
#ifndef HAVE_TDM
#define HAVE_TDM 0
#endif

// ---------------- WMMA helpers ----------------
__device__ __forceinline__ v8f wmma_bf16(v16bf a, v16bf b, v8f c) {
    return __builtin_amdgcn_wmma_f32_16x16x32_bf16(false, a, false, b, (short)0, c,
                                                   false, false);
}
// identical-to-previous WMMA sharing the A operand: set reuse-A hint
__device__ __forceinline__ v8f wmma_bf16_ra(v16bf a, v16bf b, v8f c) {
    return __builtin_amdgcn_wmma_f32_16x16x32_bf16(false, a, false, b, (short)0, c,
                                                   true, false);
}

__device__ __forceinline__ v16bf cat8(v8bf lo, v8bf hi) {
    return __builtin_shufflevector(lo, hi, 0, 1, 2, 3, 4, 5, 6, 7,
                                           8, 9, 10, 11, 12, 13, 14, 15);
}

// A 16x32 bf16 fragment from row-major [m][k]: two contiguous 8-elem runs per lane
__device__ __forceinline__ v16bf load_a_rm(const bf16_t* base, int ld, int lane) {
    int r  = lane & 15;
    int kh = (lane >> 4) * 16;                      // lanes 16..31: K+16..K+31
    const bf16_t* p = base + (size_t)r * ld + kh;
    return cat8(*(const v8bf*)p, *(const v8bf*)(p + 8));
}

// B 32x16 bf16 fragment from K-innermost layout: one contiguous 16-elem run per lane
__device__ __forceinline__ v16bf load_b_pk(const bf16_t* rowbase, int ldrow, int lane) {
    int n  = lane & 15;
    int k0 = (lane >> 4) * 16;
    return *(const v16bf*)(rowbase + (size_t)n * ldrow + k0);
}

// 16-lane-half butterfly reductions (match WMMA C-layout row striping)
__device__ __forceinline__ float half16_max(float v) {
#pragma unroll
    for (int m = 8; m >= 1; m >>= 1) v = fmaxf(v, __shfl_xor(v, m, 32));
    return v;
}
__device__ __forceinline__ float half16_sum(float v) {
#pragma unroll
    for (int m = 8; m >= 1; m >>= 1) v += __shfl_xor(v, m, 32);
    return v;
}

#if HAVE_TDM
// ---- Tensor Data Mover: 2D tile (rows x cols, 2-byte elems) global -> LDS ----
// D# per ISA ch.8: group0 = {count, lds_addr, global_addr, type=2},
// group1 = {data_size, tensor dims/strides, tile dims}; groups 2/3 zero (<=2D).
__device__ __forceinline__ void tdm_load_2d(unsigned int lds_off, const void* gaddr,
                                            unsigned int tensor_d0, unsigned int tensor_d1,
                                            unsigned long long stride0,
                                            unsigned int tile_d0, unsigned int tile_d1) {
    unsigned long long ga = (unsigned long long)gaddr;
    v4u g0;
    g0[0] = 1u;                                              // count=1 (user descriptor)
    g0[1] = lds_off;                                         // lds_addr (bytes)
    g0[2] = (unsigned int)(ga & 0xffffffffu);                // global_addr[31:0]
    g0[3] = (unsigned int)((ga >> 32) & 0x01ffffffu) | 0x80000000u;  // addr[56:32]|type=2
    v8i g1;
    g1[0] = 1 << 16;                                         // data_size=1 (2 bytes)
    g1[1] = (int)((tensor_d0 & 0xffffu) << 16);              // tensor_dim0[15:0]
    g1[2] = (int)(((tensor_d0 >> 16) & 0xffffu) | ((tensor_d1 & 0xffffu) << 16));
    g1[3] = (int)(((tensor_d1 >> 16) & 0xffffu) | ((tile_d0 & 0xffffu) << 16));
    g1[4] = (int)(tile_d1 & 0xffffu);                        // tile_dim1; tile_dim2=0
    g1[5] = (int)(unsigned int)(stride0 & 0xffffffffu);      // tensor_dim0_stride[31:0]
    g1[6] = (int)(unsigned int)((stride0 >> 32) & 0xffffu);  // stride0[47:32]; stride1=0
    g1[7] = 0;
    v4i z4 = {0, 0, 0, 0};
#if __clang_major__ >= 23
    v8i z8 = {0, 0, 0, 0, 0, 0, 0, 0};
    __builtin_amdgcn_tensor_load_to_lds(g0, g1, z4, z4, z8, 0);
#else
    __builtin_amdgcn_tensor_load_to_lds(g0, g1, z4, z4, 0);
#endif
}
#endif

// ---------------- Kernel 1: weights fp32 -> bf16 (row-major [o][c] kept) ----------------
__global__ void cvt_weights(const float* __restrict__ wqkv, const float* __restrict__ wproj,
                            bf16_t* __restrict__ wq_bf, bf16_t* __restrict__ wp_bf) {
    const int n1 = QKV_C * CC, n2 = CC * CC;
    for (int i = blockIdx.x * blockDim.x + threadIdx.x; i < n1 + n2;
         i += gridDim.x * blockDim.x) {
        if (i < n1) wq_bf[i] = (bf16_t)wqkv[i];
        else        wp_bf[i - n1] = (bf16_t)wproj[i - n1];
    }
}

// ---------------- Kernel 2: GroupNorm -> h packed [b][c/32][t][32] bf16 ----------------
__global__ void gn_kernel(const float* __restrict__ x, const float* __restrict__ gsc,
                          const float* __restrict__ gbs, bf16_t* __restrict__ hP) {
    const int b = blockIdx.x >> 5;
    const int g = blockIdx.x & 31;
    const int N = 16 * TT;
    const float* xp = x + ((size_t)b * CC + g * 16) * TT;

    float s = 0.f, ss = 0.f;
    for (int i = threadIdx.x; i < N; i += blockDim.x) {
        float v = xp[i];
        s += v; ss += v * v;
    }
#pragma unroll
    for (int m = 16; m >= 1; m >>= 1) {
        s  += __shfl_xor(s,  m, 32);
        ss += __shfl_xor(ss, m, 32);
    }
    __shared__ float sbuf[8], ssbuf[8];
    int lane = threadIdx.x & 31, wave = threadIdx.x >> 5;
    if (lane == 0) { sbuf[wave] = s; ssbuf[wave] = ss; }
    __syncthreads();
    s = 0.f; ss = 0.f;
#pragma unroll
    for (int w = 0; w < 8; ++w) { s += sbuf[w]; ss += ssbuf[w]; }
    float mu   = s * (1.0f / N);
    float var  = ss * (1.0f / N) - mu * mu;
    float rstd = rsqrtf(var + 1e-5f);

    for (int i = threadIdx.x; i < N; i += blockDim.x) {
        int c = g * 16 + (i >> 10);
        int t = i & (TT - 1);
        float v = (xp[i] - mu) * rstd * gsc[c] + gbs[c];
        hP[(((size_t)b * 16 + (c >> 5)) * TT + t) * 32 + (c & 31)] = (bf16_t)v;
    }
}

// ---------------- Kernel 3: QKV GEMM -> q̂,k̂ [bh][t][64], v̂ [bh][64][t] ----------------
// Each wave computes a 16x32 output tile: one A fragment feeds two WMMAs (reuse_a).
__global__ void qkv_gemm(const bf16_t* __restrict__ wq, const bf16_t* __restrict__ hP,
                         const float* __restrict__ bqkv, bf16_t* __restrict__ qhat,
                         bf16_t* __restrict__ khat, bf16_t* __restrict__ vhat) {
    __shared__ alignas(32) bf16_t sB[16][32][32];   // K=512 x N=32 panel, 32 KB
    const int lane = threadIdx.x & 31, wave = threadIdx.x >> 5;
    const int n0 = blockIdx.x * 32;                 // t tile (2 x 16)
    const int m0 = blockIdx.y * 128 + wave * 16;    // o tile
    const int b  = blockIdx.z;

    const bf16_t* src = hP + (size_t)b * 16 * TT * 32;
    for (int cid = threadIdx.x; cid < 2048; cid += 256) {
        int p = cid >> 7, rem = cid & 127, n = rem >> 2, q8 = (rem & 3) * 8;
        *(int4*)&sB[p][n][q8] =
            *(const int4*)(src + ((size_t)p * TT + n0 + n) * 32 + q8);
    }
    __syncthreads();

    v8f acc0 = {}, acc1 = {};
#pragma unroll
    for (int kk = 0; kk < CC; kk += 32) {
        const bf16_t* abase = wq + (size_t)m0 * CC + kk;
        __builtin_prefetch(abase + 64, 0, 1);
        v16bf af  = load_a_rm(abase, CC, lane);
        v16bf bf0 = load_b_pk(&sB[kk >> 5][0][0],  32, lane);
        v16bf bf1 = load_b_pk(&sB[kk >> 5][16][0], 32, lane);
        acc0 = wmma_bf16(af, bf0, acc0);
        acc1 = wmma_bf16_ra(af, bf1, acc1);       // same A as previous WMMA
    }

    const int rbase = (lane & 16) ? 8 : 0;
    const int o0    = m0 + rbase;                   // 8 consecutive o per lane
    const int head  = o0 / 192;
    const int which = (o0 % 192) >> 6;              // 0=q 1=k 2=v (uniform per tile)
    const int bh    = b * NHD + head;
    const int cl0   = o0 & 63;

#pragma unroll
    for (int half = 0; half < 2; ++half) {
        const v8f& acc = half ? acc1 : acc0;
        const int tcol = n0 + 16 * half + (lane & 15);
        float vr[8];
#pragma unroll
        for (int r = 0; r < 8; ++r) {
            vr[r] = acc[r] + bqkv[o0 + r];
            if (which < 2) vr[r] *= 0.35355339059327373f;   // 64^(-1/4) pre-scale q,k
        }
        if (which == 2) {                           // v̂ [bh][c][t]
#pragma unroll
            for (int r = 0; r < 8; ++r)
                vhat[((size_t)bh * CH + cl0 + r) * TT + tcol] = (bf16_t)vr[r];
        } else {                                    // q̂/k̂ [bh][t][64] : one b128 store
            v8bf pk;
#pragma unroll
            for (int r = 0; r < 8; ++r) pk[r] = (bf16_t)vr[r];
            bf16_t* dst = (which == 0 ? qhat : khat) + ((size_t)bh * TT + tcol) * CH + cl0;
            *(v8bf*)dst = pk;
        }
    }
}

// ---------------- Kernel 4: flash attention -> a packed [b][c/32][t][32] ----------------
__global__ void flash_attn(const bf16_t* __restrict__ qhat, const bf16_t* __restrict__ khat,
                           const bf16_t* __restrict__ vhat, bf16_t* __restrict__ aP) {
    __shared__ alignas(32) bf16_t sK[32][64];       // 32 keys x 64 ch, 4 KB
    __shared__ alignas(32) bf16_t sV[64][32];       // 64 ch x 32 keys, 4 KB
    __shared__ alignas(32) bf16_t sP[4][16][32];    // P tile per wave, 4 KB

    const int lane = threadIdx.x & 31, wave = threadIdx.x >> 5;   // 4 waves
    const int t0 = blockIdx.x * 64 + wave * 16;
    const int bh = blockIdx.y;
    const int b  = bh >> 3, head = bh & 7;

    const bf16_t* qb = qhat + (size_t)bh * TT * CH;
    const bf16_t* kb = khat + (size_t)bh * TT * CH;
    const bf16_t* vb = vhat + (size_t)bh * CH * TT;

    // q̂ A-fragments: loaded once, reused for all 32 key blocks
    v16bf aq0 = load_a_rm(qb + (size_t)t0 * CH,      CH, lane);
    v16bf aq1 = load_a_rm(qb + (size_t)t0 * CH + 32, CH, lane);

    v8f o_acc[4] = {{}, {}, {}, {}};
    float m_r[8], l_r[8];
#pragma unroll
    for (int r = 0; r < 8; ++r) { m_r[r] = -3.0e38f; l_r[r] = 0.f; }

#if HAVE_TDM
    const unsigned int ldsK = (unsigned int)(uintptr_t)&sK[0][0];
    const unsigned int ldsV = (unsigned int)(uintptr_t)&sV[0][0];
#endif

    for (int sb = 0; sb < TT; sb += 32) {
        __syncthreads();                            // prior reads done before overwrite
#if HAVE_TDM
        if (wave == 0) {                            // TDM: one wave DMAs both tiles
            // sK[s][c]: 32 rows of 64 contiguous ch; sV[c][s]: 64 rows of 32 keys
            tdm_load_2d(ldsK, kb + (size_t)sb * CH, CH, TT, CH, CH, 32);
            tdm_load_2d(ldsV, vb + sb, TT, CH, TT, 32, CH);
            __builtin_amdgcn_s_wait_tensorcnt(0);
        }
#else
        for (int cid = threadIdx.x; cid < 256; cid += 128) {
            int s = cid >> 3, c8 = (cid & 7) * 8;   // sK: [s][c]
            *(int4*)&sK[s][c8] = *(const int4*)(kb + ((size_t)(sb + s)) * CH + c8);
        }
        for (int cid = threadIdx.x; cid < 256; cid += 128) {
            int c = cid >> 2, s8 = (cid & 3) * 8;   // sV: [c][s]
            *(int4*)&sV[c][s8] = *(const int4*)(vb + (size_t)c * TT + sb + s8);
        }
#endif
        __syncthreads();

        // ---- S tiles 16x32 (K = 64 channels) ----
        v8f s0 = {}, s1 = {};
        {
            v16bf b0 = load_b_pk(&sK[0][0],  CH, lane);
            v16bf b1 = load_b_pk(&sK[16][0], CH, lane);
            s0 = wmma_bf16(aq0, b0, s0);
            s1 = wmma_bf16_ra(aq0, b1, s1);
            v16bf b2 = load_b_pk(&sK[0][32],  CH, lane);
            v16bf b3 = load_b_pk(&sK[16][32], CH, lane);
            s0 = wmma_bf16(aq1, b2, s0);
            s1 = wmma_bf16_ra(aq1, b3, s1);
        }

        // ---- online softmax per row; P -> LDS (bf16, A-fragment layout) ----
        const int rowbase = (lane & 16) ? 8 : 0;
#pragma unroll
        for (int r = 0; r < 8; ++r) {
            float rmax = half16_max(fmaxf(s0[r], s1[r]));
            float newm = fmaxf(m_r[r], rmax);
            float corr = __expf(m_r[r] - newm);
            float p0 = __expf(s0[r] - newm);
            float p1 = __expf(s1[r] - newm);
            float rsum = half16_sum(p0 + p1);
            l_r[r] = l_r[r] * corr + rsum;
            m_r[r] = newm;
#pragma unroll
            for (int i = 0; i < 4; ++i) o_acc[i][r] *= corr;
            int row = rowbase + r;
            sP[wave][row][lane & 15]        = (bf16_t)p0;
            sP[wave][row][16 + (lane & 15)] = (bf16_t)p1;
        }
        // same-wave LDS RAW: DS ops are in-order within a wave (no barrier needed)
        v16bf pf = load_a_rm(&sP[wave][0][0], 32, lane);

        // ---- O += P * V^T : 4 c-tiles, shared A fragment ----
        v16bf vb0 = load_b_pk(&sV[0][0],  32, lane);
        o_acc[0] = wmma_bf16(pf, vb0, o_acc[0]);
#pragma unroll
        for (int i = 1; i < 4; ++i) {
            v16bf vbi = load_b_pk(&sV[16 * i][0], 32, lane);
            o_acc[i] = wmma_bf16_ra(pf, vbi, o_acc[i]);
        }
    }

    // ---- normalize + store packed a[b][c/32][t][32] ----
    const int rowbase = (lane & 16) ? 8 : 0;
#pragma unroll
    for (int i = 0; i < 4; ++i) {
        int c = head * CH + 16 * i + (lane & 15);
#pragma unroll
        for (int r = 0; r < 8; ++r) {
            int t = t0 + rowbase + r;
            float v = o_acc[i][r] / l_r[r];
            aP[(((size_t)b * 16 + (c >> 5)) * TT + t) * 32 + (c & 31)] = (bf16_t)v;
        }
    }
}

// ---------------- Kernel 5: proj GEMM + bias + residual, f32 out ----------------
__global__ void proj_gemm(const bf16_t* __restrict__ wp, const bf16_t* __restrict__ aP,
                          const float* __restrict__ bproj, const float* __restrict__ x,
                          float* __restrict__ out) {
    __shared__ alignas(32) bf16_t sB[16][32][32];   // 32 KB
    const int lane = threadIdx.x & 31, wave = threadIdx.x >> 5;
    const int n0 = blockIdx.x * 32;
    const int m0 = blockIdx.y * 128 + wave * 16;
    const int b  = blockIdx.z;

    const bf16_t* src = aP + (size_t)b * 16 * TT * 32;
    for (int cid = threadIdx.x; cid < 2048; cid += 256) {
        int p = cid >> 7, rem = cid & 127, n = rem >> 2, q8 = (rem & 3) * 8;
        *(int4*)&sB[p][n][q8] =
            *(const int4*)(src + ((size_t)p * TT + n0 + n) * 32 + q8);
    }
    __syncthreads();

    v8f acc0 = {}, acc1 = {};
#pragma unroll
    for (int kk = 0; kk < CC; kk += 32) {
        const bf16_t* abase = wp + (size_t)m0 * CC + kk;
        __builtin_prefetch(abase + 64, 0, 1);
        v16bf af  = load_a_rm(abase, CC, lane);
        v16bf bf0 = load_b_pk(&sB[kk >> 5][0][0],  32, lane);
        v16bf bf1 = load_b_pk(&sB[kk >> 5][16][0], 32, lane);
        acc0 = wmma_bf16(af, bf0, acc0);
        acc1 = wmma_bf16_ra(af, bf1, acc1);
    }

    const int rbase = (lane & 16) ? 8 : 0;
#pragma unroll
    for (int half = 0; half < 2; ++half) {
        const v8f& acc = half ? acc1 : acc0;
        const int tcol = n0 + 16 * half + (lane & 15);
#pragma unroll
        for (int r = 0; r < 8; ++r) {
            int o = m0 + rbase + r;
            size_t idx = ((size_t)b * CC + o) * TT + tcol;
            out[idx] = x[idx] + acc[r] + bproj[o];
        }
    }
}

// ---------------- host launcher ----------------
extern "C" void kernel_launch(void* const* d_in, const int* in_sizes, int n_in,
                              void* d_out, int out_size, void* d_ws, size_t ws_size,
                              hipStream_t stream) {
    const float* x     = (const float*)d_in[0];
    const float* gsc   = (const float*)d_in[1];
    const float* gbs   = (const float*)d_in[2];
    const float* wqkv  = (const float*)d_in[3];
    const float* bqkv  = (const float*)d_in[4];
    const float* wproj = (const float*)d_in[5];
    const float* bproj = (const float*)d_in[6];
    float* out = (float*)d_out;

    // workspace carve-up (bf16): ~86 MB total, resident in the 192 MB L2
    bf16_t* wq_bf = (bf16_t*)d_ws;                       // [1536][512]
    bf16_t* wp_bf = wq_bf + (size_t)QKV_C * CC;          // [512][512]
    bf16_t* h_pk  = wp_bf + (size_t)CC * CC;             // [b][16][1024][32]
    bf16_t* qhat  = h_pk + (size_t)NB * CC * TT;         // [bh][1024][64]
    bf16_t* khat  = qhat + (size_t)NB * NHD * TT * CH;   // [bh][1024][64]
    bf16_t* vhat  = khat + (size_t)NB * NHD * TT * CH;   // [bh][64][1024]
    bf16_t* a_pk  = vhat + (size_t)NB * NHD * CH * TT;   // [b][16][1024][32]

    cvt_weights<<<1024, 256, 0, stream>>>(wqkv, wproj, wq_bf, wp_bf);
    gn_kernel<<<NB * 32, 256, 0, stream>>>(x, gsc, gbs, h_pk);
    qkv_gemm<<<dim3(TT / 32, QKV_C / 128, NB), 256, 0, stream>>>(wq_bf, h_pk, bqkv,
                                                                 qhat, khat, vhat);
    flash_attn<<<dim3(TT / 64, NB * NHD), 128, 0, stream>>>(qhat, khat, vhat, a_pk);
    proj_gemm<<<dim3(TT / 32, CC / 128, NB), 256, 0, stream>>>(wp_bf, a_pk, bproj, x, out);
}
